// GPTOSSTransformerBlock_36197984371235
// MI455X (gfx1250) — compile-verified, matrix operand order checked by reading, not run on hardware
//
#include <hip/hip_runtime.h>
#include <math.h>

// ---------------------------------------------------------------------------
// GPT-OSS transformer block for MI455X (gfx1250, wave32, WMMA).
// bf16 WMMA (16x16x32, f32 accum) for all GEMMs; sparse top-2 MoE.
// B tiles staged to LDS via gfx1250 async load-to-LDS (double buffered).
// ---------------------------------------------------------------------------

typedef __bf16 bf16;
typedef bf16 v16bf __attribute__((ext_vector_type(16)));
typedef bf16 v8bf  __attribute__((ext_vector_type(8)));
typedef float v8f  __attribute__((ext_vector_type(8)));
typedef int v4i32 __attribute__((vector_size(16)));

#define B_   2
#define S_   2048
#define D_   1024
#define H_   16
#define HKV_ 4
#define HD_  64
#define F_   2048
#define E_   8
#define T_   (B_ * S_)          // 4096 tokens
#define QN_  (H_ * HD_)         // 1024
#define KVN_ (HKV_ * HD_)       // 256

#if __has_builtin(__builtin_amdgcn_global_load_async_to_lds_b128)
#define ASYNC_LDS_PATH 1
#else
#define ASYNC_LDS_PATH 0
#endif

// ---- WMMA helpers ----------------------------------------------------------
// ISA 16-bit A/B fragment layout (16x32): lane half h = lane>>4 holds
// K elements [kbase+8h, +8) and [kbase+16+8h, +8)  (two contiguous 16B chunks).
__device__ __forceinline__ v16bf load_frag(const bf16* __restrict__ row, int kbase, int half) {
  const v8bf c0 = *reinterpret_cast<const v8bf*>(row + kbase + 8 * half);
  const v8bf c1 = *reinterpret_cast<const v8bf*>(row + kbase + 16 + 8 * half);
  v16bf f;
#pragma unroll
  for (int i = 0; i < 8; ++i) { f[i] = c0[i]; f[i + 8] = c1[i]; }
  return f;
}

__device__ __forceinline__ v8f vzero8() {
  v8f z;
#pragma unroll
  for (int i = 0; i < 8; ++i) z[i] = 0.0f;
  return z;
}

__device__ __forceinline__ v8f wmma_bf16(v16bf a, v16bf b, v8f c) {
  return __builtin_amdgcn_wmma_f32_16x16x32_bf16(false, a, false, b, (short)0, c, false, false);
}

// ---- async global -> LDS staging (16B per call) ----------------------------
__device__ __forceinline__ void stage16(const bf16* g, bf16* l) {
#if ASYNC_LDS_PATH
  __builtin_amdgcn_global_load_async_to_lds_b128(
      (__attribute__((address_space(1))) v4i32*)g,
      (__attribute__((address_space(3))) v4i32*)l, 0, 0);
#else
  *reinterpret_cast<uint4*>(l) = *reinterpret_cast<const uint4*>(g);
#endif
}

__device__ __forceinline__ void async_wait0() {
#if ASYNC_LDS_PATH
  asm volatile("s_wait_asynccnt 0x0" ::: "memory");
#endif
}

// ---- utility kernels -------------------------------------------------------
__global__ void k_zero_f32(float* __restrict__ p, long n) {
  long i = (long)blockIdx.x * 256 + threadIdx.x;
  if (i < n) p[i] = 0.0f;
}

// W[K][N] f32 (row-major) -> Wt[N][K] bf16 (transposed), so B-fragments are
// contiguous per-lane rows.
__global__ void k_cvt_transpose(const float* __restrict__ W, bf16* __restrict__ Wt,
                                int K, int N) {
  int n = blockIdx.x * 32 + (threadIdx.x & 31);
  int k = blockIdx.y * 8 + (threadIdx.x >> 5);
  if (n < N && k < K) Wt[(size_t)n * K + k] = (bf16)W[(size_t)k * N + n];
}

__global__ void k_rmsnorm1(const float* __restrict__ x, const float* __restrict__ w,
                           bf16* __restrict__ out) {
  const int t = blockIdx.x;
  const float* row = x + (size_t)t * D_;
  float s = 0.0f;
  for (int d = threadIdx.x; d < D_; d += 256) { float v = row[d]; s += v * v; }
#pragma unroll
  for (int o = 16; o >= 1; o >>= 1) s += __shfl_xor(s, o, 32);
  __shared__ float red[8];
  if ((threadIdx.x & 31) == 0) red[threadIdx.x >> 5] = s;
  __syncthreads();
  s = 0.0f;
#pragma unroll
  for (int i = 0; i < 8; ++i) s += red[i];
  const float inv = rsqrtf(s / (float)D_ + 1e-5f);
  for (int d = threadIdx.x; d < D_; d += 256)
    out[(size_t)t * D_ + d] = (bf16)(row[d] * inv * w[d]);
}

// ---- generic GEMM: C[M][N] f32 = A[M][K] bf16 x Bt[N][K] bf16 --------------
// 8 waves/block; block tile 128x64. Shared 64x32 B tile staged to LDS with
// async DMA, double-buffered; A fragments software-pipelined in registers.
__global__ void __launch_bounds__(256)
k_gemm(const bf16* __restrict__ A, const bf16* __restrict__ Bt,
       float* __restrict__ C, int M, int N, int K) {
  __shared__ __align__(16) bf16 btile[2][64 * 32];
  const int tid = threadIdx.x;
  const int lane = tid & 31, wave = tid >> 5;
  const int half = lane >> 4, l16 = lane & 15;
  const int nbase = blockIdx.x * 64;
  const int mbase = blockIdx.y * 128 + wave * 16;

  // staging role: threads 0..127 each stage one 16B chunk of the 64x32 tile
  const int srow = (tid & 127) >> 1, scol = ((tid & 127) & 1) * 16;
  const bf16* bsrc = Bt + (size_t)(nbase + srow) * K + scol;
  bf16* bdst0 = &btile[0][srow * 32 + scol];
  bf16* bdst1 = &btile[1][srow * 32 + scol];

  int ar = mbase + l16; if (ar > M - 1) ar = M - 1;
  const bf16* Arow = A + (size_t)ar * K;

  v8f acc[4] = {vzero8(), vzero8(), vzero8(), vzero8()};
  const int KT = K / 32;

  if (tid < 128) stage16(bsrc, bdst0);
  v16bf af = load_frag(Arow, 0, half);

  int buf = 0;
  for (int kt = 0; kt < KT; ++kt) {
    async_wait0();
    __syncthreads();
    if (kt + 1 < KT) {
      if (tid < 128) stage16(bsrc + (kt + 1) * 32, buf ? bdst0 : bdst1);
    }
    v16bf af_next = af;
    if (kt + 1 < KT) af_next = load_frag(Arow, (kt + 1) * 32, half);
#pragma unroll
    for (int t = 0; t < 4; ++t) {
      const v16bf bfr = load_frag(&btile[buf][(t * 16 + l16) * 32], 0, half);
      acc[t] = wmma_bf16(af, bfr, acc[t]);
    }
    af = af_next;
    buf ^= 1;
  }
#pragma unroll
  for (int t = 0; t < 4; ++t)
#pragma unroll
    for (int r = 0; r < 8; ++r) {
      const int m = mbase + r + 8 * half;
      if (m < M) C[(size_t)m * N + nbase + t * 16 + l16] = acc[t][r];
    }
}

// ---- RoPE + bf16 convert (Q scaled by 1/sqrt(HD)); V stored transposed -----
__global__ void k_rope(const float* __restrict__ qf, const float* __restrict__ kf,
                       const float* __restrict__ vf, bf16* __restrict__ qb,
                       bf16* __restrict__ kb, bf16* __restrict__ vt) {
  const int t = blockIdx.x;
  const int b = t >> 11, s = t & (S_ - 1);
  for (int p = threadIdx.x; p < H_ * 32; p += 256) {
    const int h = p >> 5, i = p & 31;
    const float ang = (float)s * __powf(10000.0f, -(float)i / 32.0f);
    float sn, c; __sincosf(ang, &sn, &c);
    const float* qr = qf + (size_t)t * QN_ + h * HD_;
    bf16* qo = qb + (size_t)t * QN_ + h * HD_;
    const float x1 = qr[i], x2 = qr[32 + i];
    qo[i]      = (bf16)(0.125f * (x1 * c - x2 * sn));
    qo[32 + i] = (bf16)(0.125f * (x2 * c + x1 * sn));
  }
  for (int p = threadIdx.x; p < HKV_ * 32; p += 256) {
    const int kvh = p >> 5, i = p & 31;
    const float ang = (float)s * __powf(10000.0f, -(float)i / 32.0f);
    float sn, c; __sincosf(ang, &sn, &c);
    const float* kr = kf + (size_t)t * KVN_ + kvh * HD_;
    bf16* ko = kb + (size_t)t * KVN_ + kvh * HD_;
    const float x1 = kr[i], x2 = kr[32 + i];
    ko[i]      = (bf16)(x1 * c - x2 * sn);
    ko[32 + i] = (bf16)(x2 * c + x1 * sn);
  }
  for (int p = threadIdx.x; p < KVN_; p += 256) {
    const int kvh = p >> 6, d = p & 63;
    vt[(((size_t)b * HKV_ + kvh) * HD_ + d) * S_ + s] = (bf16)vf[(size_t)t * KVN_ + p];
  }
}

// ---- causal flash attention (GQA), 4 waves/block, 16 q-rows/wave -----------
__global__ void __launch_bounds__(128)
k_attn(const bf16* __restrict__ qb, const bf16* __restrict__ kb,
       const bf16* __restrict__ vt, bf16* __restrict__ ctx) {
  const int lane = threadIdx.x & 31, wave = threadIdx.x >> 5;
  const int half = lane >> 4, l16 = lane & 15;
  const int b = blockIdx.z, h = blockIdx.y, kvh = h >> 2;
  const int mbase = blockIdx.x * 64 + wave * 16;
  __shared__ __align__(16) bf16 plds[4][16][32];

  const bf16* qrow = qb + ((size_t)(b * S_ + mbase + l16) * QN_ + h * HD_);
  const v16bf qf0 = load_frag(qrow, 0, half);
  const v16bf qf1 = load_frag(qrow, 32, half);

  float mx[8], sm[8];
  v8f acc[4] = {vzero8(), vzero8(), vzero8(), vzero8()};
#pragma unroll
  for (int r = 0; r < 8; ++r) { mx[r] = -1e30f; sm[r] = 0.0f; }

  const int nkb = (mbase >> 5) + 1;          // key blocks of 32 (causal bound)
  for (int kbv = 0; kbv < nkb; ++kbv) {
    const int kbbase = kbv * 32;
    const bf16* krow0 = kb + ((size_t)(b * S_ + kbbase + l16) * KVN_ + kvh * HD_);
    const bf16* krow1 = krow0 + (size_t)16 * KVN_;
    v8f s0 = vzero8(), s1 = vzero8();
    s0 = wmma_bf16(qf0, load_frag(krow0, 0, half), s0);
    s0 = wmma_bf16(qf1, load_frag(krow0, 32, half), s0);
    s1 = wmma_bf16(qf0, load_frag(krow1, 0, half), s1);
    s1 = wmma_bf16(qf1, load_frag(krow1, 32, half), s1);

#pragma unroll
    for (int r = 0; r < 8; ++r) {
      const int m = mbase + r + 8 * half;
      float e0 = (kbbase + l16      <= m) ? s0[r] : -1e30f;
      float e1 = (kbbase + 16 + l16 <= m) ? s1[r] : -1e30f;
      float rm = fmaxf(e0, e1);
#pragma unroll
      for (int o = 8; o >= 1; o >>= 1) rm = fmaxf(rm, __shfl_xor(rm, o, 16));
      const float nm = fmaxf(mx[r], rm);
      const float scale = __expf(mx[r] - nm);
      const float p0 = __expf(e0 - nm), p1 = __expf(e1 - nm);
      float rs = p0 + p1;
#pragma unroll
      for (int o = 8; o >= 1; o >>= 1) rs += __shfl_xor(rs, o, 16);
      sm[r] = sm[r] * scale + rs;
      mx[r] = nm;
#pragma unroll
      for (int t = 0; t < 4; ++t) acc[t][r] *= scale;
      plds[wave][r + 8 * half][l16]      = (bf16)p0;   // transpose P via LDS
      plds[wave][r + 8 * half][16 + l16] = (bf16)p1;
    }
    asm volatile("s_wait_dscnt 0" ::: "memory");        // LDS stores visible
    const v16bf pf = load_frag(&plds[wave][l16][0], 0, half);
#pragma unroll
    for (int t = 0; t < 4; ++t) {
      const bf16* vdim = vt + (((size_t)(b * HKV_ + kvh) * HD_) + t * 16 + l16) * S_;
      acc[t] = wmma_bf16(pf, load_frag(vdim, kbbase, half), acc[t]);
    }
  }
#pragma unroll
  for (int t = 0; t < 4; ++t)
#pragma unroll
    for (int r = 0; r < 8; ++r) {
      const int m = mbase + r + 8 * half;
      ctx[(size_t)(b * S_ + m) * QN_ + h * HD_ + t * 16 + l16] = (bf16)(acc[t][r] / sm[r]);
    }
}

// ---- residual + clip + RMSNorm2 -------------------------------------------
__global__ void k_resid_norm(const float* __restrict__ hs, const float* __restrict__ ao,
                             const float* __restrict__ w, float* __restrict__ resid,
                             bf16* __restrict__ hn) {
  const int t = blockIdx.x;
  float vloc[4];
  float s = 0.0f;
  int idx = 0;
  for (int d = threadIdx.x; d < D_; d += 256, ++idx) {
    float v = hs[(size_t)t * D_ + d] + ao[(size_t)t * D_ + d];
    v = fminf(100.0f, fmaxf(-100.0f, v));
    resid[(size_t)t * D_ + d] = v;
    vloc[idx] = v;
    s += v * v;
  }
#pragma unroll
  for (int o = 16; o >= 1; o >>= 1) s += __shfl_xor(s, o, 32);
  __shared__ float red[8];
  if ((threadIdx.x & 31) == 0) red[threadIdx.x >> 5] = s;
  __syncthreads();
  s = 0.0f;
#pragma unroll
  for (int i = 0; i < 8; ++i) s += red[i];
  const float inv = rsqrtf(s / (float)D_ + 1e-5f);
  idx = 0;
  for (int d = threadIdx.x; d < D_; d += 256, ++idx)
    hn[(size_t)t * D_ + d] = (bf16)(vloc[idx] * inv * w[d]);
}

// ---- router: logits, softmax, top-2, renorm --------------------------------
__global__ void k_router(const bf16* __restrict__ hn2, const float* __restrict__ Wr,
                         float* __restrict__ topw, int* __restrict__ topi,
                         int* __restrict__ counts) {
  const int lane = threadIdx.x & 31, wave = threadIdx.x >> 5;
  const int t = blockIdx.x * 8 + wave;
  const bf16* row = hn2 + (size_t)t * D_;
  float acc[E_];
#pragma unroll
  for (int e = 0; e < E_; ++e) acc[e] = 0.0f;
  for (int d = lane; d < D_; d += 32) {
    const float hv = (float)row[d];
    const float* wr = Wr + (size_t)d * E_;
#pragma unroll
    for (int e = 0; e < E_; ++e) acc[e] += hv * wr[e];
  }
#pragma unroll
  for (int e = 0; e < E_; ++e)
#pragma unroll
    for (int o = 16; o >= 1; o >>= 1) acc[e] += __shfl_xor(acc[e], o, 32);
  if (lane == 0) {
    float mx = acc[0];
    for (int e = 1; e < E_; ++e) mx = fmaxf(mx, acc[e]);
    float ex[E_], ssum = 0.0f;
    for (int e = 0; e < E_; ++e) { ex[e] = __expf(acc[e] - mx); ssum += ex[e]; }
    int i0 = 0;
    for (int e = 1; e < E_; ++e) if (acc[e] > acc[i0]) i0 = e;
    int i1 = (i0 == 0) ? 1 : 0;
    for (int e = 0; e < E_; ++e) if (e != i0 && acc[e] > acc[i1]) i1 = e;
    const float p0 = ex[i0] / ssum, p1 = ex[i1] / ssum;
    const float wsum = p0 + p1;
    topi[2 * t] = i0; topi[2 * t + 1] = i1;
    topw[2 * t] = p0 / wsum; topw[2 * t + 1] = p1 / wsum;
    atomicAdd(&counts[i0], 1);
    atomicAdd(&counts[i1], 1);
  }
}

__global__ void k_offsets(const int* __restrict__ counts, int* __restrict__ offs,
                          int* __restrict__ cursor) {
  if (threadIdx.x == 0 && blockIdx.x == 0) {
    int o = 0;
    for (int e = 0; e < E_; ++e) { offs[e] = o; cursor[e] = o; o += counts[e]; }
  }
}

__global__ void k_scatter(const int* __restrict__ topi, const float* __restrict__ topw,
                          int* __restrict__ cursor, unsigned* __restrict__ slot_tok,
                          float* __restrict__ slot_w) {
  const int t = blockIdx.x * 256 + threadIdx.x;
  if (t >= T_) return;
#pragma unroll
  for (int j = 0; j < 2; ++j) {
    const int e = topi[2 * t + j];
    const int pos = atomicAdd(&cursor[e], 1);
    slot_tok[pos] = (unsigned)t;
    slot_w[pos] = topw[2 * t + j];
  }
}

// ---- MoE GEMM1: gathered rows x (WgT,WuT), fused silu(g)*u -> abuf ---------
__global__ void __launch_bounds__(256)
k_moe_gemm1(const bf16* __restrict__ hn2, const bf16* __restrict__ WgT,
            const bf16* __restrict__ WuT, const unsigned* __restrict__ slot_tok,
            const int* __restrict__ offs, const int* __restrict__ counts,
            bf16* __restrict__ abuf) {
  __shared__ __align__(16) bf16 gtile[2][64 * 32];
  __shared__ __align__(16) bf16 utile[2][64 * 32];
  const int e = blockIdx.y >> 5, mb = blockIdx.y & 31;
  const int cnt = counts[e];
  if (mb * 128 >= cnt) return;                       // block-uniform exit
  const int tid = threadIdx.x;
  const int lane = tid & 31, wave = tid >> 5;
  const int half = lane >> 4, l16 = lane & 15;
  const int mbase = mb * 128 + wave * 16;
  const int off0 = offs[e];
  const int nbase = blockIdx.x * 64;

  const int srow = (tid & 127) >> 1, scol = ((tid & 127) & 1) * 16;
  const bf16* gsrc = WgT + ((size_t)e * F_ + nbase + srow) * D_ + scol;
  const bf16* usrc = WuT + ((size_t)e * F_ + nbase + srow) * D_ + scol;
  bf16* gdst[2] = {&gtile[0][srow * 32 + scol], &gtile[1][srow * 32 + scol]};
  bf16* udst[2] = {&utile[0][srow * 32 + scol], &utile[1][srow * 32 + scol]};

  int mr = mbase + l16; if (mr > cnt - 1) mr = cnt - 1;
  const bf16* Arow = hn2 + (size_t)slot_tok[off0 + mr] * D_;

  v8f g[4] = {vzero8(), vzero8(), vzero8(), vzero8()};
  v8f u[4] = {vzero8(), vzero8(), vzero8(), vzero8()};
  const int KT = D_ / 32;

  if (tid < 128) stage16(gsrc, gdst[0]);
  else           stage16(usrc, udst[0]);
  v16bf af = load_frag(Arow, 0, half);

  int buf = 0;
  for (int kt = 0; kt < KT; ++kt) {
    async_wait0();
    __syncthreads();
    if (kt + 1 < KT) {
      if (tid < 128) stage16(gsrc + (kt + 1) * 32, gdst[buf ^ 1]);
      else           stage16(usrc + (kt + 1) * 32, udst[buf ^ 1]);
    }
    v16bf af_next = af;
    if (kt + 1 < KT) af_next = load_frag(Arow, (kt + 1) * 32, half);
#pragma unroll
    for (int t = 0; t < 4; ++t) {
      g[t] = wmma_bf16(af, load_frag(&gtile[buf][(t * 16 + l16) * 32], 0, half), g[t]);
      u[t] = wmma_bf16(af, load_frag(&utile[buf][(t * 16 + l16) * 32], 0, half), u[t]);
    }
    af = af_next;
    buf ^= 1;
  }
#pragma unroll
  for (int t = 0; t < 4; ++t)
#pragma unroll
    for (int r = 0; r < 8; ++r) {
      const int m = mbase + r + 8 * half;
      if (m < cnt) {
        const float gg = g[t][r];
        const float a = gg / (1.0f + __expf(-gg)) * u[t][r];
        abuf[(size_t)(off0 + m) * F_ + nbase + t * 16 + l16] = (bf16)a;
      }
    }
}

// ---- MoE GEMM2: abuf x WdT, weighted atomic scatter-add into moe_out -------
__global__ void __launch_bounds__(256)
k_moe_gemm2(const bf16* __restrict__ abuf, const bf16* __restrict__ WdT,
            const unsigned* __restrict__ slot_tok, const float* __restrict__ slot_w,
            const int* __restrict__ offs, const int* __restrict__ counts,
            float* __restrict__ moe) {
  __shared__ __align__(16) bf16 btile[2][64 * 32];
  const int e = blockIdx.y >> 5, mb = blockIdx.y & 31;
  const int cnt = counts[e];
  if (mb * 128 >= cnt) return;                       // block-uniform exit
  const int tid = threadIdx.x;
  const int lane = tid & 31, wave = tid >> 5;
  const int half = lane >> 4, l16 = lane & 15;
  const int mbase = mb * 128 + wave * 16;
  const int off0 = offs[e];
  const int nbase = blockIdx.x * 64;

  const int srow = (tid & 127) >> 1, scol = ((tid & 127) & 1) * 16;
  const bf16* bsrc = WdT + ((size_t)e * D_ + nbase + srow) * F_ + scol;
  bf16* bdst0 = &btile[0][srow * 32 + scol];
  bf16* bdst1 = &btile[1][srow * 32 + scol];

  int mr = mbase + l16; if (mr > cnt - 1) mr = cnt - 1;
  const bf16* Arow = abuf + (size_t)(off0 + mr) * F_;

  v8f acc[4] = {vzero8(), vzero8(), vzero8(), vzero8()};
  const int KT = F_ / 32;

  if (tid < 128) stage16(bsrc, bdst0);
  v16bf af = load_frag(Arow, 0, half);

  int buf = 0;
  for (int kt = 0; kt < KT; ++kt) {
    async_wait0();
    __syncthreads();
    if (kt + 1 < KT) {
      if (tid < 128) stage16(bsrc + (kt + 1) * 32, buf ? bdst0 : bdst1);
    }
    v16bf af_next = af;
    if (kt + 1 < KT) af_next = load_frag(Arow, (kt + 1) * 32, half);
#pragma unroll
    for (int t = 0; t < 4; ++t)
      acc[t] = wmma_bf16(af, load_frag(&btile[buf][(t * 16 + l16) * 32], 0, half), acc[t]);
    af = af_next;
    buf ^= 1;
  }
#pragma unroll
  for (int r = 0; r < 8; ++r) {
    const int m = mbase + r + 8 * half;
    if (m < cnt) {
      const int slot = off0 + m;
      const unsigned tok = slot_tok[slot];
      const float w = slot_w[slot];
#pragma unroll
      for (int t = 0; t < 4; ++t)
        atomicAdd(&moe[(size_t)tok * D_ + nbase + t * 16 + l16], acc[t][r] * w);
    }
  }
}

__global__ void k_final(const float* __restrict__ resid, const float* __restrict__ moe,
                        float* __restrict__ out, long n) {
  const long i = (long)blockIdx.x * 256 + threadIdx.x;
  if (i < n) {
    const float v = resid[i] + moe[i];
    out[i] = fminf(100.0f, fmaxf(-100.0f, v));
  }
}

// ---------------------------------------------------------------------------
extern "C" void kernel_launch(void* const* d_in, const int* in_sizes, int n_in,
                              void* d_out, int out_size, void* d_ws, size_t ws_size,
                              hipStream_t stream) {
  (void)in_sizes; (void)n_in; (void)out_size; (void)ws_size;
  const float* hs  = (const float*)d_in[0];
  const float* ln1 = (const float*)d_in[1];
  const float* ln2 = (const float*)d_in[2];
  const float* Wq  = (const float*)d_in[3];
  const float* Wk  = (const float*)d_in[4];
  const float* Wv  = (const float*)d_in[5];
  const float* Wo  = (const float*)d_in[6];
  const float* Wr  = (const float*)d_in[7];
  const float* Wg  = (const float*)d_in[8];
  const float* Wu  = (const float*)d_in[9];
  const float* Wd  = (const float*)d_in[10];
  float* out = (float*)d_out;

  char* base = (char*)d_ws;
  size_t off = 0;
  auto take = [&](size_t bytes) -> char* {
    char* p = base + off;
    off = (off + bytes + 255) & ~(size_t)255;
    return p;
  };

  bf16*  hn1   = (bf16*)take((size_t)T_ * D_ * 2);     // reused as ctx after QKV
  float* qf    = (float*)take((size_t)T_ * QN_ * 4);   // reused as attn_out
  float* kf    = (float*)take((size_t)T_ * KVN_ * 4);
  float* vf    = (float*)take((size_t)T_ * KVN_ * 4);
  bf16*  qb    = (bf16*)take((size_t)T_ * QN_ * 2);
  bf16*  kbuf  = (bf16*)take((size_t)T_ * KVN_ * 2);
  bf16*  vtb   = (bf16*)take((size_t)T_ * KVN_ * 2);
  float* resid = (float*)take((size_t)T_ * D_ * 4);
  bf16*  hn2   = (bf16*)take((size_t)T_ * D_ * 2);
  bf16*  WqT   = (bf16*)take((size_t)QN_ * D_ * 2);
  bf16*  WkT   = (bf16*)take((size_t)KVN_ * D_ * 2);
  bf16*  WvT   = (bf16*)take((size_t)KVN_ * D_ * 2);
  bf16*  WoT   = (bf16*)take((size_t)D_ * QN_ * 2);
  bf16*  WgT   = (bf16*)take((size_t)E_ * F_ * D_ * 2);
  bf16*  WuT   = (bf16*)take((size_t)E_ * F_ * D_ * 2);
  bf16*  WdT   = (bf16*)take((size_t)E_ * D_ * F_ * 2);
  bf16*  abuf  = (bf16*)take((size_t)2 * T_ * F_ * 2);
  float* moe   = (float*)take((size_t)T_ * D_ * 4);
  int*   topi  = (int*)take((size_t)T_ * 2 * 4);
  float* topw  = (float*)take((size_t)T_ * 2 * 4);
  int*   counts = (int*)take(64);
  int*   offs   = (int*)take(64);
  int*   cursor = (int*)take(64);
  unsigned* slot_tok = (unsigned*)take((size_t)2 * T_ * 4);
  float*    slot_w   = (float*)take((size_t)2 * T_ * 4);

  bf16*  ctxb     = hn1;   // hn1 dead after QKV GEMMs
  float* attn_out = qf;    // qf dead after RoPE

  const dim3 blk256(256);

  // 1) weights -> bf16 transposed (one pass; bf16 set (~100MB) then L2-resident)
  k_cvt_transpose<<<dim3(QN_ / 32, D_ / 8), blk256, 0, stream>>>(Wq, WqT, D_, QN_);
  k_cvt_transpose<<<dim3(KVN_ / 32, D_ / 8), blk256, 0, stream>>>(Wk, WkT, D_, KVN_);
  k_cvt_transpose<<<dim3(KVN_ / 32, D_ / 8), blk256, 0, stream>>>(Wv, WvT, D_, KVN_);
  k_cvt_transpose<<<dim3(D_ / 32, QN_ / 8), blk256, 0, stream>>>(Wo, WoT, QN_, D_);
  for (int e = 0; e < E_; ++e) {
    k_cvt_transpose<<<dim3(F_ / 32, D_ / 8), blk256, 0, stream>>>(
        Wg + (size_t)e * D_ * F_, WgT + (size_t)e * F_ * D_, D_, F_);
    k_cvt_transpose<<<dim3(F_ / 32, D_ / 8), blk256, 0, stream>>>(
        Wu + (size_t)e * D_ * F_, WuT + (size_t)e * F_ * D_, D_, F_);
    k_cvt_transpose<<<dim3(D_ / 32, F_ / 8), blk256, 0, stream>>>(
        Wd + (size_t)e * F_ * D_, WdT + (size_t)e * D_ * F_, F_, D_);
  }

  // 2) RMSNorm1 + QKV GEMMs + RoPE
  k_rmsnorm1<<<T_, blk256, 0, stream>>>(hs, ln1, hn1);
  k_gemm<<<dim3(QN_ / 64, T_ / 128), blk256, 0, stream>>>(hn1, WqT, qf, T_, QN_, D_);
  k_gemm<<<dim3(KVN_ / 64, T_ / 128), blk256, 0, stream>>>(hn1, WkT, kf, T_, KVN_, D_);
  k_gemm<<<dim3(KVN_ / 64, T_ / 128), blk256, 0, stream>>>(hn1, WvT, vf, T_, KVN_, D_);
  k_rope<<<T_, blk256, 0, stream>>>(qf, kf, vf, qb, kbuf, vtb);

  // 3) causal flash attention (WMMA QK^T and P.V)
  k_attn<<<dim3(S_ / 64, H_, B_), dim3(128), 0, stream>>>(qb, kbuf, vtb, ctxb);

  // 4) output projection + residual + clip + RMSNorm2
  k_gemm<<<dim3(D_ / 64, T_ / 128), blk256, 0, stream>>>(ctxb, WoT, attn_out, T_, D_, QN_);
  k_resid_norm<<<T_, blk256, 0, stream>>>(hs, attn_out, ln2, resid, hn2);

  // 5) router + compaction (sparse top-2 MoE: 1/4 of the dense FLOPs)
  k_zero_f32<<<((long)T_ * D_ + 255) / 256, blk256, 0, stream>>>(moe, (long)T_ * D_);
  k_zero_f32<<<1, blk256, 0, stream>>>((float*)counts, 8);
  k_router<<<T_ / 8, blk256, 0, stream>>>(hn2, Wr, topw, topi, counts);
  k_offsets<<<1, 1, 0, stream>>>(counts, offs, cursor);
  k_scatter<<<(T_ + 255) / 256, blk256, 0, stream>>>(topi, topw, cursor, slot_tok, slot_w);

  // 6) expert GEMMs over gathered token lists
  k_moe_gemm1<<<dim3(F_ / 64, E_ * 32), blk256, 0, stream>>>(hn2, WgT, WuT, slot_tok,
                                                             offs, counts, abuf);
  k_moe_gemm2<<<dim3(D_ / 64, E_ * 32), blk256, 0, stream>>>(abuf, WdT, slot_tok, slot_w,
                                                             offs, counts, moe);

  // 7) final residual + clip
  k_final<<<((long)T_ * D_ + 255) / 256, blk256, 0, stream>>>(resid, moe, out, (long)T_ * D_);
}
